// VqVae_81423989998112
// MI455X (gfx1250) — matrix operand order repaired and checked by reading, not compiled, not run on hardware
//
#include <hip/hip_runtime.h>

// MI455X / gfx1250 fused VQ-VAE forward.
// f16 WMMA (v_wmma_f32_16x16x32_f16) for all GEMMs + VQ distance matmuls.
// Weights/codebooks are staged per-block into LDS in pre-swizzled B-fragment
// layout (coalesced f32 reads -> f16), so each WMMA consumes 2x ds_load_b128
// for B instead of 16 scalar global loads.

typedef __attribute__((ext_vector_type(16))) _Float16 v16h;
typedef __attribute__((ext_vector_type(8)))  _Float16 v8h;
typedef __attribute__((ext_vector_type(8)))  float    v8f;

#define TA    90      // T*A
#define TAP   96      // padded K for first layer (3 chunks of 32)
#define HID   128
#define DLAT  512
#define GQ    4
#define KCODE 32
#define ROWS  64      // rows per block tile (4 stripes x 16, wave pairs)
#define ZSTR  520     // halfs; 1040B rows -> +4 banks/row, conflict-free A loads
#define BSTR  136     // halfs; 272B rows  -> +4 banks/row
#define FRAGH 512     // halfs per staged 32x16 B fragment

union F16x16 { v16h v; v8h h[2]; _Float16 e[16]; };

static __device__ __forceinline__ v8f wmma_f16(v16h a, v16h b, v8f c) {
  return __builtin_amdgcn_wmma_f32_16x16x32_f16(false, a, false, b, (short)0, c,
                                                false, false);
}

// A fragment (16x32 f16) from LDS activations, rows mBase..mBase+15.
// lane L: M = L%16 ; element e -> K = kBase + e + 16*(L/16)
static __device__ __forceinline__ v16h ldsA(const _Float16* in, int stride,
                                            int mBase, int kBase, int lane) {
  const int hi = lane >> 4;
  const _Float16* row = in + (mBase + (lane & 15)) * stride;
  F16x16 a;
  a.h[0] = *(const v8h*)(row + kBase + 16 * hi);
  a.h[1] = *(const v8h*)(row + kBase + 16 * hi + 8);
  return a.v;
}

// B fragment from staged LDS: frag-major, per-lane chunks of 8 halfs at
// [frag*512 + lane*8] and [frag*512 + 256 + lane*8]  (both 16B aligned).
static __device__ __forceinline__ v16h ldsB(const _Float16* ws, int frag,
                                            int lane) {
  const _Float16* base = ws + (size_t)frag * FRAGH + lane * 8;
  F16x16 b;
  b.h[0] = *(const v8h*)(base);
  b.h[1] = *(const v8h*)(base + 256);
  return b.v;
}

// Stage weights [Kreal x Nreal] f32 (row-major) -> LDS B-fragments f16,
// zero-padded to [Kpad x Nd]. Coalesced reads (consecutive tid -> consecutive n).
static __device__ void stage_w(const float* __restrict__ w, _Float16* ws,
                               int Kpad, int Kreal, int Nd, int Nreal, int tid) {
  const int nK = Kpad >> 5;
  const int total = Kpad * Nd;
  for (int e = tid; e < total; e += 256) {
    const int n = e % Nd;
    const int k = e / Nd;
    const float f = (k < Kreal && n < Nreal) ? w[(size_t)k * Nreal + n] : 0.0f;
    const int kc = k >> 5, kr = k & 31, hi = kr >> 4, i = kr & 15;
    const int ntI = n >> 4, nl = n & 15, ln = (hi << 4) | nl;
    ws[(size_t)(ntI * nK + kc) * FRAGH + ((i >> 3) << 8) + ln * 8 + (i & 7)] =
        (_Float16)f;
  }
}

// Stage one codebook group [32 codes x 512 dims] as transposed B (K=dim, N=code).
// Consecutive tid -> consecutive dim -> coalesced reads of cb rows.
static __device__ void stage_cb(const float* __restrict__ cb, _Float16* ws,
                                int tid) {
  const int nK = DLAT >> 5;  // 16
  for (int e = tid; e < DLAT * KCODE; e += 256) {
    const int d = e & (DLAT - 1);
    const int c = e >> 9;
    const float f = cb[(size_t)c * DLAT + d];
    const int kc = d >> 5, kr = d & 31, hi = kr >> 4, i = kr & 15;
    const int ntI = c >> 4, nl = c & 15, ln = (hi << 4) | nl;
    ws[(size_t)(ntI * nK + kc) * FRAGH + ((i >> 3) << 8) + ln * 8 + (i & 7)] =
        (_Float16)f;
  }
}

// Dense layer from staged weights. Wave pair splits n-tiles (nOff=0/16, step 32).
static __device__ void layer_lds(const _Float16* in, int inStr, _Float16* out,
                                 int outStr, const _Float16* ws,
                                 const float* __restrict__ bias, int Kpad,
                                 int Nd, bool relu, int mBase, int nOff,
                                 int lane) {
  const int nK = Kpad >> 5;
  for (int nt = nOff; nt < Nd; nt += 32) {
    v8f c = {};
    const int fBase = (nt >> 4) * nK;
    for (int kc = 0; kc < nK; ++kc) {
      v16h a = ldsA(in, inStr, mBase, kc << 5, lane);
      v16h b = ldsB(ws, fBase + kc, lane);
      c = wmma_f16(a, b, c);
    }
    const int n    = nt + (lane & 15);
    const float bb = bias[n];
    const int mHi  = mBase + ((lane >> 4) << 3);
#pragma unroll
    for (int r = 0; r < 8; ++r) {
      float v = c[r] + bb;
      if (relu) v = fmaxf(v, 0.0f);
      out[(mHi + r) * outStr + n] = (_Float16)v;
    }
  }
}

// Output layer -> global f32 [N x Nreal] (Nd padded, columns >= Nreal dropped).
static __device__ void layer_out(const _Float16* in, int inStr,
                                 const _Float16* ws,
                                 const float* __restrict__ bias, int Kpad,
                                 int Nd, int Nreal, float* __restrict__ outG,
                                 int gRow0, int mBase, int nOff, int lane) {
  const int nK = Kpad >> 5;
  for (int nt = nOff; nt < Nd; nt += 32) {
    v8f c = {};
    const int fBase = (nt >> 4) * nK;
    for (int kc = 0; kc < nK; ++kc) {
      v16h a = ldsA(in, inStr, mBase, kc << 5, lane);
      v16h b = ldsB(ws, fBase + kc, lane);
      c = wmma_f16(a, b, c);
    }
    const int n = nt + (lane & 15);
    if (n < Nreal) {
      const float bb = bias[n];
      const int mHi  = mBase + ((lane >> 4) << 3);
#pragma unroll
      for (int r = 0; r < 8; ++r)
        outG[(size_t)(gRow0 + mHi + r) * Nreal + n] = c[r] + bb;
    }
  }
}

__global__ void vqvae_init_loss(float* p) { *p = 0.0f; }

__global__ __launch_bounds__(256, 1)
void vqvae_fused(const float* __restrict__ state,
                 const float* __restrict__ ew1, const float* __restrict__ eb1,
                 const float* __restrict__ ew2, const float* __restrict__ eb2,
                 const float* __restrict__ ew3, const float* __restrict__ eb3,
                 const float* __restrict__ dw1, const float* __restrict__ db1,
                 const float* __restrict__ dw2, const float* __restrict__ db2,
                 const float* __restrict__ dw3, const float* __restrict__ db3,
                 const float* __restrict__ cbs,
                 float* __restrict__ outQuant, float* __restrict__ outCode,
                 float* __restrict__ outRecon, float* __restrict__ outLoss,
                 int N) {
  extern __shared__ char smem[];
  _Float16* bufZ   = (_Float16*)smem;                 // residual z [64][520]
  _Float16* bufQ   = bufZ + ROWS * ZSTR;              // quant f16  [64][520]
  _Float16* buf1   = bufQ;                            // h-buffer aliases bufQ head
  _Float16* buf0   = bufQ + ROWS * ZSTR;              // h-buffer   [64][136]
  _Float16* wstage = buf0 + ROWS * BSTR;              // 65536 halfs (128KB)
  float*    enorm  = (float*)(wstage + DLAT * HID);   // [G*K]
  int*      codes  = (int*)(enorm + GQ * KCODE);      // [64][G]
  float*    lred   = (float*)(codes + ROWS * GQ);     // [8]

  const int tid   = threadIdx.x;
  const int lane  = tid & 31;
  const int wave  = tid >> 5;
  const int mBase = (wave >> 1) * 16;            // stripe of the wave pair
  const int nOff  = (wave & 1) * 16;             // n-tile interleave in pair
  const int row0  = blockIdx.x * ROWS;           // N assumed multiple of 64

  // ---- stage x tile (f16, zero-pad cols 90..95), enc1 weights, code norms
  for (int e = tid; e < ROWS * TAP; e += 256) {
    const int r = e / TAP, c = e % TAP;
    buf0[r * BSTR + c] =
        (_Float16)((c < TA) ? state[(size_t)(row0 + r) * TA + c] : 0.0f);
  }
  stage_w(ew1, wstage, TAP, TA, HID, HID, tid);
  if (tid < GQ * KCODE) {
    const float* e = cbs + (size_t)tid * DLAT;
    float s = 0.0f;
    for (int d = 0; d < DLAT; ++d) { const float t = e[d]; s += t * t; }
    enorm[tid] = s;
  }
  __syncthreads();

  // ---- encoder: 90->128 relu, 128->128 relu, 128->512
  layer_lds(buf0, BSTR, buf1, BSTR, wstage, eb1, TAP, HID, true, mBase, nOff, lane);
  __syncthreads();
  stage_w(ew2, wstage, HID, HID, HID, HID, tid);
  __syncthreads();
  layer_lds(buf1, BSTR, buf0, BSTR, wstage, eb2, HID, HID, true, mBase, nOff, lane);
  __syncthreads();
  stage_w(ew3, wstage, HID, HID, DLAT, DLAT, tid);
  __syncthreads();
  layer_lds(buf0, BSTR, bufZ, ZSTR, wstage, eb3, HID, DLAT, false, mBase, nOff, lane);
  __syncthreads();

  // ---- residual VQ: per group, stage codebook as B-frags, WMMA distances,
  //      wave-local argmin (duplicated in the pair), dim-split residual update.
  float lossAcc = 0.0f;
  const int hi = lane >> 4;
  const int nl = lane & 15;
  const int dBase = ((wave & 1) << 8) + lane * 8;   // pair splits 512 dims
  for (int g = 0; g < GQ; ++g) {
    const float* cbg = cbs + (size_t)g * KCODE * DLAT;
    stage_cb(cbg, wstage, tid);
    __syncthreads();
    v8f c0 = {}, c1 = {};
    for (int kc = 0; kc < (DLAT >> 5); ++kc) {
      v16h a = ldsA(bufZ, ZSTR, mBase, kc << 5, lane);
      c0 = wmma_f16(a, ldsB(wstage, kc, lane), c0);                // codes 0..15
      c1 = wmma_f16(a, ldsB(wstage, (DLAT >> 5) + kc, lane), c1);  // codes 16..31
    }
    const float en0 = enorm[g * KCODE + nl];
    const float en1 = enorm[g * KCODE + 16 + nl];
    int idxArr[8];
#pragma unroll
    for (int r = 0; r < 8; ++r) {                 // row m = mBase + 8*hi + r
      const float s0 = en0 - 2.0f * c0[r];
      const float s1 = en1 - 2.0f * c1[r];
      float v; int ix;
      if (s1 < s0) { v = s1; ix = 16 + nl; } else { v = s0; ix = nl; }
#pragma unroll
      for (int mm = 1; mm < 16; mm <<= 1) {       // argmin across 16-lane half
        const float vv = __shfl_xor(v, mm, 32);
        const int   ii = __shfl_xor(ix, mm, 32);
        if (vv < v || (vv == v && ii < ix)) { v = vv; ix = ii; }
      }
      idxArr[r] = ix;
    }
    __syncthreads();   // all bufZ/wstage reads done before updates / restage
#pragma unroll 1
    for (int ro = 0; ro < 16; ++ro) {
      const int ix = __shfl(idxArr[ro & 7], (ro >> 3) << 4, 32);
      if (lane == 0 && (wave & 1) == 0) codes[(mBase + ro) * GQ + g] = ix;
      const float* e = cbg + (size_t)ix * DLAT + dBase;
      _Float16* zr = bufZ + (mBase + ro) * ZSTR + dBase;
#pragma unroll
      for (int j = 0; j < 8; ++j) {
        const float ev = e[j];
        const float rv = (float)zr[j];
        const float df = ev - rv;                 // q - residual
        lossAcc += df * df;
        zr[j] = (_Float16)(rv - ev);              // residual -= q
      }
    }
  }
  __syncthreads();

  // ---- quant = sum of gathered codes (exact straight-through algebra)
#pragma unroll 1
  for (int ro = 0; ro < 16; ++ro) {
    const int row = mBase + ro;
    const size_t grow = (size_t)(row0 + row);
    if (lane < GQ && (wave & 1) == 0)
      outCode[grow * GQ + lane] = (float)codes[row * GQ + lane];
    const float* e0 = cbs + (size_t)(0 * KCODE + codes[row * GQ + 0]) * DLAT + dBase;
    const float* e1 = cbs + (size_t)(1 * KCODE + codes[row * GQ + 1]) * DLAT + dBase;
    const float* e2 = cbs + (size_t)(2 * KCODE + codes[row * GQ + 2]) * DLAT + dBase;
    const float* e3 = cbs + (size_t)(3 * KCODE + codes[row * GQ + 3]) * DLAT + dBase;
    _Float16* qr = bufQ + row * ZSTR + dBase;
    float* qg = outQuant + grow * DLAT + dBase;
#pragma unroll
    for (int j = 0; j < 8; ++j) {
      const float s = e0[j] + e1[j] + e2[j] + e3[j];
      qg[j] = s;
      qr[j] = (_Float16)s;
    }
  }
  stage_w(dw1, wstage, DLAT, DLAT, HID, HID, tid);
  __syncthreads();

  // ---- decoder: 512->128 relu, 128->128 relu, 128->90
  layer_lds(bufQ, ZSTR, buf0, BSTR, wstage, db1, DLAT, HID, true, mBase, nOff, lane);
  __syncthreads();   // bufQ reads done before buf1 (alias) is overwritten
  stage_w(dw2, wstage, HID, HID, HID, HID, tid);
  __syncthreads();
  layer_lds(buf0, BSTR, buf1, BSTR, wstage, db2, HID, HID, true, mBase, nOff, lane);
  __syncthreads();
  stage_w(dw3, wstage, HID, HID, TAP, TA, tid);
  __syncthreads();
  layer_out(buf1, BSTR, wstage, db3, HID, TAP, TA, outRecon, row0, mBase, nOff, lane);

  // ---- vq_loss reduction: sum over groups of mean((q-r)^2)
#pragma unroll
  for (int mm = 1; mm < 32; mm <<= 1) lossAcc += __shfl_xor(lossAcc, mm, 32);
  if (lane == 0) lred[wave] = lossAcc;
  __syncthreads();
  if (tid == 0) {
    float s = 0.0f;
    for (int w = 0; w < 8; ++w) s += lred[w];
    atomicAdd(outLoss, s * (1.0f / ((float)N * (float)DLAT)));
  }
}

extern "C" void kernel_launch(void* const* d_in, const int* in_sizes, int n_in,
                              void* d_out, int out_size, void* d_ws, size_t ws_size,
                              hipStream_t stream) {
  const float* state = (const float*)d_in[0];
  const float* ew1 = (const float*)d_in[1];  const float* eb1 = (const float*)d_in[2];
  const float* ew2 = (const float*)d_in[3];  const float* eb2 = (const float*)d_in[4];
  const float* ew3 = (const float*)d_in[5];  const float* eb3 = (const float*)d_in[6];
  const float* dw1 = (const float*)d_in[7];  const float* db1 = (const float*)d_in[8];
  const float* dw2 = (const float*)d_in[9];  const float* db2 = (const float*)d_in[10];
  const float* dw3 = (const float*)d_in[11]; const float* db3 = (const float*)d_in[12];
  const float* cbs = (const float*)d_in[13];

  const int N = in_sizes[0] / TA;              // 131072 (multiple of ROWS)
  float* out      = (float*)d_out;
  float* outQuant = out;
  float* outCode  = out + (size_t)N * DLAT;
  float* outRecon = outCode + (size_t)N * GQ;
  float* outLoss  = outRecon + (size_t)N * TA;

  const size_t shmem =
      (size_t)2 * ROWS * ZSTR * sizeof(_Float16) +   // bufZ + bufQ(+buf1 alias)
      (size_t)ROWS * BSTR * sizeof(_Float16) +       // buf0
      (size_t)DLAT * HID * sizeof(_Float16) +        // wstage (128KB)
      (size_t)GQ * KCODE * sizeof(float) +           // enorm
      (size_t)ROWS * GQ * sizeof(int) +              // codes
      (size_t)8 * sizeof(float);                     // lred  (~283 KB < 320 KB)

  (void)hipFuncSetAttribute((const void*)vqvae_fused,
                            hipFuncAttributeMaxDynamicSharedMemorySize,
                            (int)shmem);

  vqvae_init_loss<<<1, 1, 0, stream>>>(outLoss);
  vqvae_fused<<<N / ROWS, 256, shmem, stream>>>(
      state, ew1, eb1, ew2, eb2, ew3, eb3, dw1, db1, dw2, db2, dw3, db3, cbs,
      outQuant, outCode, outRecon, outLoss, N);
}
